// GraphFractionalSDELayer_18098992185603
// MI455X (gfx1250) — compile-verified, matrix operand order checked by reading, not run on hardware
//
#include <hip/hip_runtime.h>
#include <hip/hip_bf16.h>
#include <cmath>

// ---------------------------------------------------------------------------
// GraphFractionalSDELayer for MI455X (gfx1250, wave32, WMMA)
// GEMMs: v_wmma_f32_16x16x32_bf16, LDS-staged tiles (async-to-LDS builtin
// confirmed present in this toolchain), fused epilogues.
// ---------------------------------------------------------------------------

typedef __attribute__((ext_vector_type(16))) __bf16 v16bf;
typedef __attribute__((ext_vector_type(8)))  float  v8f;
typedef int vi4_t __attribute__((vector_size(16)));   // type expected by async builtin

// Problem constants (fixed by the reference)
#define BN      32768          // B*N nodes
#define HH      256            // D == H == O == 256
#define SZ      (BN * HH)      // elements per activation tensor = 8388608
#define NEDGE   131072
#define NSTEPS  10
#define KP      264            // padded LDS row stride (elements) -> bank spread

// ------------------------------- helpers -----------------------------------

__device__ __forceinline__ float bf2f_lo(unsigned u) { return __builtin_bit_cast(float, u << 16); }
__device__ __forceinline__ float bf2f_hi(unsigned u) { return __builtin_bit_cast(float, u & 0xFFFF0000u); }

__device__ __forceinline__ unsigned short f2bf_bits(float f) {
    unsigned u = __builtin_bit_cast(unsigned, f);
    u += 0x7FFFu + ((u >> 16) & 1u);          // round-to-nearest-even
    return (unsigned short)(u >> 16);
}
__device__ __forceinline__ __bf16 f2bf(float f) {
    unsigned short h = f2bf_bits(f);
    return __builtin_bit_cast(__bf16, h);
}
__device__ __forceinline__ unsigned pack_bf16x2(float lo, float hi) {
    return (unsigned)f2bf_bits(lo) | ((unsigned)f2bf_bits(hi) << 16);
}

__device__ __forceinline__ unsigned rotl32(unsigned x, unsigned n) {
    return (x << n) | (x >> (32u - n));
}

// threefry2x32, 20 rounds — deterministic counter-based noise
__device__ __forceinline__ uint2 threefry2x32(uint2 k, uint2 c) {
    const unsigned rot[8] = {13u, 15u, 26u, 6u, 17u, 29u, 16u, 24u};
    unsigned ks0 = k.x, ks1 = k.y, ks2 = k.x ^ k.y ^ 0x1BD11BDAu;
    unsigned x0 = c.x + ks0, x1 = c.y + ks1;
    unsigned kk[3] = {ks0, ks1, ks2};
#pragma unroll
    for (int g = 0; g < 5; ++g) {
#pragma unroll
        for (int r = 0; r < 4; ++r) {
            x0 += x1;
            x1 = rotl32(x1, rot[(g & 1) * 4 + r]);
            x1 ^= x0;
        }
        x0 += kk[(g + 1) % 3];
        x1 += kk[(g + 2) % 3] + (unsigned)(g + 1);
    }
    uint2 out; out.x = x0; out.y = x1; return out;
}

__device__ __forceinline__ float rand_normal(unsigned step, unsigned idx) {
    uint2 key; key.x = 0x9E3779B9u ^ (step * 0x85EBCA6Bu); key.y = 42u;
    uint2 ctr; ctr.x = idx; ctr.y = 0xCAFEF00Du;
    uint2 r = threefry2x32(key, ctr);
    float u1 = (float)((r.x >> 8) + 1u) * (1.0f / 16777217.0f);   // (0,1]
    float u2 = (float)(r.y >> 8)        * (1.0f / 16777216.0f);   // [0,1)
    float rad = sqrtf(-2.0f * __logf(u1));
    return rad * __cosf(6.28318530717958647f * u2);
}

__device__ __forceinline__ float softplus_f(float x) {
    return (x > 20.0f) ? x : log1pf(__expf(x));
}

// 16-byte global->LDS copy: async tensor path if available, else load+ds_store
__device__ __forceinline__ void cp16_g2l(const __bf16* __restrict__ g,
                                         __bf16* __restrict__ l) {
#if defined(__has_builtin)
#if __has_builtin(__builtin_amdgcn_global_load_async_to_lds_b128)
    __builtin_amdgcn_global_load_async_to_lds_b128(
        (vi4_t*)g, (vi4_t*)l, 0, 0);
    return;
#else
    *(uint4*)l = *(const uint4*)g;
    return;
#endif
#else
    *(uint4*)l = *(const uint4*)g;
#endif
}

__device__ __forceinline__ void wait_async_copies() {
#if defined(__has_builtin)
#if __has_builtin(__builtin_amdgcn_s_wait_asynccnt)
    __builtin_amdgcn_s_wait_asynccnt(0);
#else
    asm volatile("s_wait_asynccnt 0x0" ::: "memory");
#endif
#else
    asm volatile("s_wait_asynccnt 0x0" ::: "memory");
#endif
    asm volatile("" ::: "memory");
}

// 32B fragment load from LDS with 16B alignment (2 x ds_load_b128)
__device__ __forceinline__ v16bf ld_frag_lds(const __bf16* p) {
    struct V { uint4 a, b; } v;
    v.a = *(const uint4*)p;
    v.b = *(const uint4*)(p + 8);
    return __builtin_bit_cast(v16bf, v);
}

// --------------------------- small utility kernels --------------------------

// Wt[n*256 + k] = bf16(W[k*256 + n])   (weights become N-major for B-fragments)
__global__ void weight_transpose_kernel(const float* __restrict__ W,
                                        __bf16* __restrict__ Wt) {
    int t = blockIdx.x * blockDim.x + threadIdx.x;   // 65536 threads
    int k = t >> 8, n = t & 255;
    Wt[n * 256 + k] = f2bf(W[k * 256 + n]);
}

// vectorized f32 -> bf16 (count multiple of 8)
__global__ void cvt_f32_to_bf16_kernel(const float* __restrict__ src,
                                       unsigned* __restrict__ dst, int count8) {
    int t = blockIdx.x * blockDim.x + threadIdx.x;
    if (t >= count8) return;
    const float4 a = ((const float4*)src)[2 * t];
    const float4 b = ((const float4*)src)[2 * t + 1];
    uint4 o;
    o.x = pack_bf16x2(a.x, a.y);
    o.y = pack_bf16x2(a.z, a.w);
    o.z = pack_bf16x2(b.x, b.y);
    o.w = pack_bf16x2(b.z, b.w);
    ((uint4*)dst)[t] = o;
}

__global__ void zero_f32_kernel(float* __restrict__ p, int count4) {
    int t = blockIdx.x * blockDim.x + threadIdx.x;
    if (t < count4) ((float4*)p)[t] = float4{0.f, 0.f, 0.f, 0.f};
}

// Weighted scatter-sum over edges: agg[dst] += h[src] * w   (f32 atomics)
__global__ void edge_scatter_kernel(const int* __restrict__ edge_index,
                                    const float* __restrict__ edge_weight,
                                    const __bf16* __restrict__ h,
                                    float* __restrict__ agg) {
    int gid = blockIdx.x * blockDim.x + threadIdx.x;   // NEDGE * 32 threads
    int e  = gid >> 5;
    int f0 = (gid & 31) * 8;
    int src = edge_index[e];
    int dst = edge_index[NEDGE + e];
    float w = edge_weight[e];
    uint4 hv = *(const uint4*)(h + (size_t)src * HH + f0);
    float* ap = agg + (size_t)dst * HH + f0;
    float v[8] = {bf2f_lo(hv.x), bf2f_hi(hv.x), bf2f_lo(hv.y), bf2f_hi(hv.y),
                  bf2f_lo(hv.z), bf2f_hi(hv.z), bf2f_lo(hv.w), bf2f_hi(hv.w)};
#pragma unroll
    for (int k = 0; k < 8; ++k) {
        __hip_atomic_fetch_add(ap + k, v[k] * w,
                               __ATOMIC_RELAXED, __HIP_MEMORY_SCOPE_AGENT);
    }
}

// Fractional-history convolution (vectorized by 8):
//   state = spatial + scale * sum_{k<nhist} w[nhist-1-k]*(drift[k]+diff[k])
// with ALPHA = 0.5 => w[j] = sqrt(j+1) - sqrt(j)
__global__ void history_combine_kernel(const float* __restrict__ spatial,
                                       const __bf16* __restrict__ driftH,
                                       const __bf16* __restrict__ diffH,
                                       int nhist, float scale,
                                       float* __restrict__ stateF,
                                       unsigned* __restrict__ stateB) {
    int g = blockIdx.x * blockDim.x + threadIdx.x;    // SZ/8 threads
    if (g >= SZ / 8) return;
    float acc[8] = {0.f, 0.f, 0.f, 0.f, 0.f, 0.f, 0.f, 0.f};
    for (int k = 0; k < nhist; ++k) {
        int j = nhist - 1 - k;
        float wj = sqrtf((float)(j + 1)) - sqrtf((float)j);
        size_t base = (size_t)k * (SZ / 8) + g;
        uint4 d = ((const uint4*)driftH)[base];
        uint4 f = ((const uint4*)diffH)[base];
        acc[0] += wj * (bf2f_lo(d.x) + bf2f_lo(f.x));
        acc[1] += wj * (bf2f_hi(d.x) + bf2f_hi(f.x));
        acc[2] += wj * (bf2f_lo(d.y) + bf2f_lo(f.y));
        acc[3] += wj * (bf2f_hi(d.y) + bf2f_hi(f.y));
        acc[4] += wj * (bf2f_lo(d.z) + bf2f_lo(f.z));
        acc[5] += wj * (bf2f_hi(d.z) + bf2f_hi(f.z));
        acc[6] += wj * (bf2f_lo(d.w) + bf2f_lo(f.w));
        acc[7] += wj * (bf2f_hi(d.w) + bf2f_hi(f.w));
    }
    float4 s0 = ((const float4*)spatial)[2 * g];
    float4 s1 = ((const float4*)spatial)[2 * g + 1];
    float o[8] = {s0.x + scale * acc[0], s0.y + scale * acc[1],
                  s0.z + scale * acc[2], s0.w + scale * acc[3],
                  s1.x + scale * acc[4], s1.y + scale * acc[5],
                  s1.z + scale * acc[6], s1.w + scale * acc[7]};
    ((float4*)stateF)[2 * g]     = float4{o[0], o[1], o[2], o[3]};
    ((float4*)stateF)[2 * g + 1] = float4{o[4], o[5], o[6], o[7]};
    uint4 ob;
    ob.x = pack_bf16x2(o[0], o[1]);
    ob.y = pack_bf16x2(o[2], o[3]);
    ob.z = pack_bf16x2(o[4], o[5]);
    ob.w = pack_bf16x2(o[6], o[7]);
    ((uint4*)stateB)[g] = ob;
}

// ------------------------------ WMMA GEMM -----------------------------------
// C(32768x256,f32-acc) = A(bf16, MxK row-major) x B(bf16, given as Bt = N-major)
// block: 256 thr = 8 waves; block tile 64M x 128N; wave tile 16M x 64N.
// Whole-K LDS staging: sA = 64x256 (padded rows), sB = 128x256 (padded rows).

enum {
    EPI_BIAS_BF16 = 0,   // outB = bf16(acc + bias[n])
    EPI_TANH_BF16 = 1,   // outB = bf16(tanh(acc + bias[n]))
    EPI_SPNOISE   = 2,   // outB = bf16(softplus(acc+bias) * N(step,idx) * sqrt_dt)
    EPI_BIAS_F32  = 3,   // outF = acc + bias[n]
    EPI_COUPLE    = 4,   // outB = bf16(0.5*(acc + bias + other1[idx]))
    EPI_ADD_F32   = 5,   // outF = acc + bias + other1[idx]
    EPI_COMBINE   = 6    // outB = bf16(0.5*(acc + bias + other1[idx] + other2[idx]))
};

template <int EPI>
__global__ __launch_bounds__(256)
void gemm_bf16_kernel(const __bf16* __restrict__ A,
                      const __bf16* __restrict__ Bt,
                      const float*  __restrict__ bias,
                      float*        __restrict__ outF,
                      __bf16*       __restrict__ outB,
                      const float*  __restrict__ other1,
                      const float*  __restrict__ other2,
                      int step, float sqrt_dt) {
    __shared__ __bf16 sA[64 * KP];    // ~33.8 KB
    __shared__ __bf16 sB[128 * KP];   // ~67.6 KB

    const int tid  = threadIdx.x;
    const int lane = tid & 31;
    const int wv   = tid >> 5;
    const int half = lane >> 4;
    const int r    = lane & 15;

    const int mBase = blockIdx.x * 64;    // gridDim.x = 512
    const int nBase = blockIdx.y * 128;   // gridDim.y = 2

    // ---- cooperative stage (16B chunks; 32 chunks per 256-elem row) ----
    const __bf16* Ablk = A  + (size_t)mBase * HH;
    const __bf16* Bblk = Bt + (size_t)nBase * HH;
#pragma unroll
    for (int c = tid; c < 64 * 32; c += 256) {
        int row = c >> 5, ch = c & 31;
        cp16_g2l(Ablk + row * HH + ch * 8, &sA[row * KP + ch * 8]);
    }
#pragma unroll
    for (int c = tid; c < 128 * 32; c += 256) {
        int row = c >> 5, ch = c & 31;
        cp16_g2l(Bblk + row * HH + ch * 8, &sB[row * KP + ch * 8]);
    }
    wait_async_copies();
    __syncthreads();

    // ---- compute out of LDS ----
    const int m0 = mBase + (wv & 3) * 16;
    const int n0 = nBase + (wv >> 2) * 64;
    const __bf16* sa = &sA[((wv & 3) * 16 + r) * KP + half * 16];
    const __bf16* sb = &sB[((wv >> 2) * 64 + r) * KP + half * 16];

    v8f c0 = {}, c1 = {}, c2 = {}, c3 = {};
#pragma unroll
    for (int kb = 0; kb < HH; kb += 32) {
        v16bf a  = ld_frag_lds(sa + kb);
        v16bf b0 = ld_frag_lds(sb + kb);
        v16bf b1 = ld_frag_lds(sb + 16 * KP + kb);
        v16bf b2 = ld_frag_lds(sb + 32 * KP + kb);
        v16bf b3 = ld_frag_lds(sb + 48 * KP + kb);
        c0 = __builtin_amdgcn_wmma_f32_16x16x32_bf16(false, a, false, b0, (short)0, c0, false, false);
        c1 = __builtin_amdgcn_wmma_f32_16x16x32_bf16(false, a, false, b1, (short)0, c1, false, false);
        c2 = __builtin_amdgcn_wmma_f32_16x16x32_bf16(false, a, false, b2, (short)0, c2, false, false);
        c3 = __builtin_amdgcn_wmma_f32_16x16x32_bf16(false, a, false, b3, (short)0, c3, false, false);
    }

    v8f* frag[4] = {&c0, &c1, &c2, &c3};
#pragma unroll
    for (int j = 0; j < 4; ++j) {
        const int n  = n0 + j * 16 + r;
        const float bv = bias[n];
        v8f c = *frag[j];
#pragma unroll
        for (int rr = 0; rr < 8; ++rr) {
            const int m = m0 + rr + 8 * half;
            const size_t idx = (size_t)m * HH + n;
            float v = c[rr] + bv;
            if constexpr (EPI == EPI_BIAS_BF16) {
                outB[idx] = f2bf(v);
            } else if constexpr (EPI == EPI_TANH_BF16) {
                outB[idx] = f2bf(tanhf(v));
            } else if constexpr (EPI == EPI_SPNOISE) {
                float nz = rand_normal((unsigned)step, (unsigned)idx);
                outB[idx] = f2bf(softplus_f(v) * nz * sqrt_dt);
            } else if constexpr (EPI == EPI_BIAS_F32) {
                outF[idx] = v;
            } else if constexpr (EPI == EPI_COUPLE) {
                outB[idx] = f2bf(0.5f * (v + other1[idx]));
            } else if constexpr (EPI == EPI_ADD_F32) {
                outF[idx] = v + other1[idx];
            } else if constexpr (EPI == EPI_COMBINE) {
                outB[idx] = f2bf(0.5f * (v + other1[idx] + other2[idx]));
            }
        }
    }
}

// ------------------------------- launcher -----------------------------------

extern "C" void kernel_launch(void* const* d_in, const int* in_sizes, int n_in,
                              void* d_out, int out_size, void* d_ws, size_t ws_size,
                              hipStream_t stream) {
    // Inputs (setup_inputs order)
    const float* x  = (const float*)d_in[0];
    const int*   ei = (const int*)  d_in[1];
    const float* ew = (const float*)d_in[2];
    const float* W[9];
    const float* bv[9];
    for (int i = 0; i < 9; ++i) {
        W[i]  = (const float*)d_in[3 + 2 * i];
        bv[i] = (const float*)d_in[4 + 2 * i];
    }
    float* out = (float*)d_out;

    // Workspace carve-up (256B aligned)
    char* ws = (char*)d_ws;
    size_t off = 0;
    auto carve = [&](size_t bytes) -> void* {
        void* p = ws + off;
        off += (bytes + 255) & ~(size_t)255;
        return p;
    };
    __bf16* Wt[9];
    for (int i = 0; i < 9; ++i) Wt[i] = (__bf16*)carve(256 * 256 * sizeof(__bf16));
    __bf16* x_bf        = (__bf16*)carve((size_t)SZ * 2);
    __bf16* h_bf        = (__bf16*)carve((size_t)SZ * 2);
    float*  spatialF    = (float*) carve((size_t)SZ * 4);
    __bf16* spatial_bf  = (__bf16*)carve((size_t)SZ * 2);
    float*  stateF      = (float*) carve((size_t)SZ * 4);
    __bf16* state_bf    = (__bf16*)carve((size_t)SZ * 2);
    __bf16* t1_bf       = (__bf16*)carve((size_t)SZ * 2);
    __bf16* driftH      = (__bf16*)carve((size_t)NSTEPS * SZ * 2);
    __bf16* diffH       = (__bf16*)carve((size_t)NSTEPS * SZ * 2);
    float*  sembF       = (float*) carve((size_t)SZ * 4);
    float*  ctF         = (float*) carve((size_t)SZ * 4);
    __bf16* coupled_bf  = (__bf16*)carve((size_t)SZ * 2);
    __bf16* combined_bf = (__bf16*)carve((size_t)SZ * 2);

    const float sqrt_dt = sqrtf(0.1f);
    const float scale   = (float)(pow(0.1, 0.5) / tgamma(1.5));   // dt^alpha / Gamma(alpha+1)

    const dim3 gemmGrid(512, 2);
    const int  T = 256;
    const int  vecBlocks = (SZ / 8) / T;   // 8-wide elementwise kernels

    // 0) weights -> bf16, N-major
    for (int i = 0; i < 9; ++i)
        weight_transpose_kernel<<<256, T, 0, stream>>>(W[i], Wt[i]);

    // 1) x -> bf16 ; h = x @ W_g + b_g
    cvt_f32_to_bf16_kernel<<<vecBlocks, T, 0, stream>>>(x, (unsigned*)x_bf, SZ / 8);
    gemm_bf16_kernel<EPI_BIAS_BF16><<<gemmGrid, T, 0, stream>>>(
        x_bf, Wt[0], bv[0], nullptr, h_bf, nullptr, nullptr, 0, 0.f);

    // 2) spatial = segment_sum(h[src] * w, dst)
    zero_f32_kernel<<<(SZ / 4) / T, T, 0, stream>>>(spatialF, SZ / 4);
    edge_scatter_kernel<<<(NEDGE * 32) / T, T, 0, stream>>>(ei, ew, h_bf, spatialF);
    cvt_f32_to_bf16_kernel<<<vecBlocks, T, 0, stream>>>(spatialF, (unsigned*)spatial_bf, SZ / 8);

    // 3) fractional SDE loop
    const __bf16* curState = spatial_bf;
    for (int i = 0; i < NSTEPS; ++i) {
        gemm_bf16_kernel<EPI_TANH_BF16><<<gemmGrid, T, 0, stream>>>(
            curState, Wt[1], bv[1], nullptr, t1_bf, nullptr, nullptr, 0, 0.f);
        gemm_bf16_kernel<EPI_BIAS_BF16><<<gemmGrid, T, 0, stream>>>(
            t1_bf, Wt[2], bv[2], nullptr, driftH + (size_t)i * SZ, nullptr, nullptr, 0, 0.f);
        gemm_bf16_kernel<EPI_SPNOISE><<<gemmGrid, T, 0, stream>>>(
            curState, Wt[3], bv[3], nullptr, diffH + (size_t)i * SZ, nullptr, nullptr, i, sqrt_dt);
        history_combine_kernel<<<vecBlocks, T, 0, stream>>>(
            spatialF, driftH, diffH, i + 1, scale, stateF, (unsigned*)state_bf);
        curState = state_bf;
    }

    // 4) coupling
    gemm_bf16_kernel<EPI_BIAS_F32><<<gemmGrid, T, 0, stream>>>(
        spatial_bf, Wt[4], bv[4], sembF, nullptr, nullptr, nullptr, 0, 0.f);
    gemm_bf16_kernel<EPI_COUPLE><<<gemmGrid, T, 0, stream>>>(
        state_bf, Wt[5], bv[5], nullptr, coupled_bf, sembF, nullptr, 0, 0.f);
    gemm_bf16_kernel<EPI_ADD_F32><<<gemmGrid, T, 0, stream>>>(
        coupled_bf, Wt[6], bv[6], ctF, nullptr, stateF, nullptr, 0, 0.f);
    gemm_bf16_kernel<EPI_COMBINE><<<gemmGrid, T, 0, stream>>>(
        coupled_bf, Wt[7], bv[7], nullptr, combined_bf, spatialF, ctF, 0, 0.f);

    // 5) output projection -> d_out (f32)
    gemm_bf16_kernel<EPI_BIAS_F32><<<gemmGrid, T, 0, stream>>>(
        combined_bf, Wt[8], bv[8], out, nullptr, nullptr, nullptr, 0, 0.f);

    (void)in_sizes; (void)n_in; (void)out_size; (void)ws_size;
}
